// SaliencySampler_84335977824812
// MI455X (gfx1250) — compile-verified, matrix operand order checked by reading, not run on hardware
//
#include <hip/hip_runtime.h>
#include <hip/hip_bf16.h>

typedef __attribute__((ext_vector_type(16))) _Float16 v16h;
typedef __attribute__((ext_vector_type(8)))  float    v8f;

#if defined(__HIP_DEVICE_COMPILE__) && __has_builtin(__builtin_amdgcn_global_load_async_to_lds_b32) && __has_builtin(__builtin_amdgcn_s_wait_asynccnt)
#define HAVE_ASYNC_LDS 1
typedef __attribute__((address_space(1))) int g_int;   // global (AS1)
typedef __attribute__((address_space(3))) int l_int;   // LDS (AS3)
#else
#define HAVE_ASYNC_LDS 0
#endif

__device__ __forceinline__ int iclamp(int v, int lo, int hi){ return v < lo ? lo : (v > hi ? hi : v); }

// ---------------------------------------------------------------------------
// K0: build the WMMA B-fragment image of sal_w ONCE (64x27 -> 4 N-tiles of
// 16x16x32 f16 in the ISA 7.12.2 16-bit layout). 4 KB total; the hot conv
// kernel then does straight 32B vector loads instead of 33M scalar loads.
// ---------------------------------------------------------------------------
__global__ __launch_bounds__(256) void k_prep_b(const float* __restrict__ sal_w,
                                                _Float16* __restrict__ bfrag){
  int idx = blockIdx.x * 256 + threadIdx.x;          // 2048 entries
  if (idx >= 2048) return;
  int ntile = idx >> 9, rem = idx & 511;
  int l = rem >> 4, e = rem & 15;
  int nn = ntile*16 + (l & 15);
  int K  = ((l >= 16) ? 8 : 0) + ((e < 8) ? e : e + 8);
  bfrag[idx] = (K < 27) ? (_Float16)sal_w[nn*27 + K] : (_Float16)0.0f;
}

// ---------------------------------------------------------------------------
// K1: transpose [B,C,T,H,W]->[BT,C,H,W] fused with 2x2 mean pool (256->128),
//     converting to f16 for the WMMA conv. Pure bandwidth.
// ---------------------------------------------------------------------------
__global__ __launch_bounds__(256) void k_pool(const float* __restrict__ x,
                                              _Float16* __restrict__ xlow){
  int o = blockIdx.x * 256 + threadIdx.x;            // ((n*3+c)*128+y)*128+x
  int xx = o & 127; int tmp = o >> 7;
  int yy = tmp & 127; tmp >>= 7;
  int c = tmp % 3; int n = tmp / 3;
  int b = n >> 5, t = n & 31;
  const float* p = x + ((((size_t)(b*3 + c))*32 + t)*256 + 2*yy)*256 + 2*xx;
  float s = 0.25f * (p[0] + p[1] + p[256] + p[257]);
  xlow[o] = (_Float16)s;
}

// ---------------------------------------------------------------------------
// K2: fused conv3x3(3->64,pad1) + bias + ReLU + conv1x1(64->1) + bias.
// Implicit GEMM: M = 16 pixels/wave, N = 64 (4 tiles), K = 27 padded to 32.
// Block = 256 threads (8 waves) = TWO image rows of 128 pixels (4 staged
// rows, halo overlap 2x instead of 3x). 8 v_wmma_f32_16x16x32_f16 per wave;
// channel reduction via shfl_xor so the 536 MB feat tensor never exists.
// ---------------------------------------------------------------------------
__global__ __launch_bounds__(256) void k_conv_wmma(const _Float16* __restrict__ xlow,
    const _Float16* __restrict__ bfrag,
    const float* __restrict__ sal_b,
    const float* __restrict__ cl_w,  const float* __restrict__ cl_b,
    float* __restrict__ xs){
  __shared__ _Float16 sIn[3*4*132];                  // 3 ch x rows py0-1..py0+2 x cols[-1..128]
  const int tid  = threadIdx.x;
  const int lane = tid & 31, wave = tid >> 5;
  const int n   = blockIdx.x >> 6;                   // image 0..127
  const int py0 = (blockIdx.x & 63) * 2;             // output rows py0, py0+1

  // Stage input rows py0-1..py0+2, 3 channels, cols -1..128 (zero conv-pad).
  for (int it = tid; it < 1560; it += 256){
    int ic = it / 520, rr = it % 520;
    int ky = rr / 130, cc = rr % 130;
    int col = cc - 1, gy = py0 + ky - 1;
    _Float16 v = (_Float16)0.0f;
    if (gy >= 0 && gy < 128 && col >= 0 && col < 128)
      v = xlow[(((size_t)n*3 + ic)*128 + gy)*128 + col];
    sIn[(ic*4 + ky)*132 + cc] = v;
  }

  // B fragments: precomputed, contiguous 32B per lane per tile.
  v16h bf[4];
  #pragma unroll
  for (int t4 = 0; t4 < 4; ++t4)
    bf[t4] = *(const v16h*)(bfrag + ((size_t)t4*32 + lane)*16);

  __syncthreads();

  // A fragments (ISA 7.12.2 16-bit A 16x32): lane holds M = lane%16 = pixel,
  // element e -> K = (lane>=16?8:0) + (e<8 ? e : e+8); K = ic*9+ky*3+kx.
  const int m  = lane & 15;
  const int px = wave*16 + m;
  v8f acc[2][4];
  #pragma unroll
  for (int r = 0; r < 2; ++r){
    v16h a;
    #pragma unroll
    for (int e = 0; e < 16; ++e){
      int K = ((lane >= 16) ? 8 : 0) + ((e < 8) ? e : e + 8);
      _Float16 v = (_Float16)0.0f;
      if (K < 27){
        int ic = K/9, r9 = K%9, ky = r9/3, kx = r9%3;
        v = sIn[(ic*4 + ky + r)*132 + (px + kx)];    // col+1 == px+kx, in-range
      }
      a[e] = v;
    }
    #pragma unroll
    for (int t4 = 0; t4 < 4; ++t4){
      v8f c = {};
      acc[r][t4] = __builtin_amdgcn_wmma_f32_16x16x32_f16(false, a, false, bf[t4],
                                                          (short)0, c, false, false);
    }
  }

  // bias + ReLU + 1x1 conv: s(m) = sum_n cl_w[n]*relu(feat[m][n]+sal_b[n]).
  float salb[4], clw[4];
  #pragma unroll
  for (int t4 = 0; t4 < 4; ++t4){
    int nn = t4*16 + m;
    salb[t4] = sal_b[nn];
    clw[t4]  = cl_w[nn];
  }
  const float clb = cl_b[0];
  #pragma unroll
  for (int r = 0; r < 2; ++r){
    float* orow = xs + ((size_t)n*128 + py0 + r)*128 + wave*16;
    #pragma unroll
    for (int v = 0; v < 8; ++v){
      float s = 0.0f;
      #pragma unroll
      for (int t4 = 0; t4 < 4; ++t4)
        s += clw[t4] * fmaxf(acc[r][t4][v] + salb[t4], 0.0f);
      s += __shfl_xor(s, 1, 32);                     // reduce the 16-lane halves:
      s += __shfl_xor(s, 2, 32);                     // lanes 0-15  -> M=v
      s += __shfl_xor(s, 4, 32);                     // lanes 16-31 -> M=v+8
      s += __shfl_xor(s, 8, 32);
      if (m == 0)
        orow[v + ((lane >> 4) << 3)] = s + clb;
    }
  }
}

// ---------------------------------------------------------------------------
// K3: bilinear resize 128->32 (align_corners) + softmax over 1024, one block
// per image, LDS tree reductions.
// ---------------------------------------------------------------------------
__global__ __launch_bounds__(256) void k_resize_softmax(const float* __restrict__ xs,
                                                        float* __restrict__ hm){
  __shared__ float red[256];
  const int n = blockIdx.x, tid = threadIdx.x;
  const float* src = xs + (size_t)n*128*128;
  const float sc = 127.0f/31.0f;
  float v[4];
  #pragma unroll
  for (int k = 0; k < 4; ++k){
    int pos = tid + k*256;
    int oy = pos >> 5, ox = pos & 31;
    float fy = oy*sc, fx = ox*sc;
    int y0 = (int)fy, x0 = (int)fx;
    int y1 = min(y0+1, 127), x1 = min(x0+1, 127);
    float wy = fy - y0, wx = fx - x0;
    float a = src[y0*128+x0], b = src[y0*128+x1];
    float c = src[y1*128+x0], d = src[y1*128+x1];
    v[k] = (a*(1.0f-wx) + b*wx)*(1.0f-wy) + (c*(1.0f-wx) + d*wx)*wy;
  }
  float lm = fmaxf(fmaxf(v[0],v[1]), fmaxf(v[2],v[3]));
  red[tid] = lm; __syncthreads();
  for (int s = 128; s > 0; s >>= 1){ if (tid < s) red[tid] = fmaxf(red[tid], red[tid+s]); __syncthreads(); }
  float mx = red[0]; __syncthreads();
  float e[4], ls = 0.0f;
  #pragma unroll
  for (int k = 0; k < 4; ++k){ e[k] = __expf(v[k]-mx); ls += e[k]; }
  red[tid] = ls; __syncthreads();
  for (int s = 128; s > 0; s >>= 1){ if (tid < s) red[tid] += red[tid+s]; __syncthreads(); }
  float inv = 1.0f/red[0];
  float* dst = hm + (size_t)n*1024;
  #pragma unroll
  for (int k = 0; k < 4; ++k) dst[tid + k*256] = e[k]*inv;
}

// ---------------------------------------------------------------------------
// K4: separable 65-tap gaussian on the edge-padded 96x96 heatmap -> 32x32
// sampling grid. Edge replication => only 32 distinct padded rows/cols, so
// both passes run over 32x32 in LDS. LDS fill uses the gfx1250 async
// global->LDS path (ASYNCcnt) when the builtin is available.
// ---------------------------------------------------------------------------
__global__ __launch_bounds__(256) void k_gauss_grid(const float* __restrict__ hm,
    const float* __restrict__ gauss, float* __restrict__ grid32){
  __shared__ float sHm[1024], sH0[1024], sH1[1024], sG[65];
  const int n = blockIdx.x, tid = threadIdx.x;
#if HAVE_ASYNC_LDS
  #pragma unroll
  for (int k = 0; k < 4; ++k){
    int i = tid + k*256;
    __builtin_amdgcn_global_load_async_to_lds_b32(
        (g_int*)(hm + (size_t)n*1024 + i),
        (l_int*)&sHm[i], 0, 0);
  }
  if (tid < 65)
    __builtin_amdgcn_global_load_async_to_lds_b32(
        (g_int*)(gauss + 32*65 + tid),
        (l_int*)&sG[tid], 0, 0);
  __builtin_amdgcn_s_wait_asynccnt(0);
#else
  if (tid < 65) sG[tid] = gauss[32*65 + tid];        // g(i,j)=g1(i)g1(j), g1(32)=1
  for (int i = tid; i < 1024; i += 256) sHm[i] = hm[(size_t)n*1024 + i];
#endif
  __syncthreads();
  // Horizontal pass over the 32 distinct rows.
  for (int it = tid; it < 1024; it += 256){
    int r = it >> 5, ox = it & 31;
    const float* row = &sHm[r*32];
    float s0 = 0.0f, s1 = 0.0f;
    for (int j = 0; j < 65; ++j){
      int colp = ox + j;                             // padded col 0..95
      float hv = row[iclamp(colp - 32, 0, 31)];
      float g  = sG[j];
      s0 += g*hv;
      s1 += g*hv*((float)(colp - 32)*(1.0f/31.0f));  // Px at padded position
    }
    sH0[it] = s0; sH1[it] = s1;
  }
  __syncthreads();
  for (int it = tid; it < 1024; it += 256){
    int oy = it >> 5, ox = it & 31;
    float pf = 0.0f, ax = 0.0f, ay = 0.0f;
    for (int i = 0; i < 65; ++i){
      int rowp = oy + i;
      int rr = iclamp(rowp - 32, 0, 31);
      float g  = sG[i];
      float h0 = sH0[rr*32 + ox];
      pf += g*h0;
      ax += g*sH1[rr*32 + ox];
      ay += g*h0*((float)(rowp - 32)*(1.0f/31.0f));  // Py at padded position
    }
    float gx = fminf(fmaxf(ax/pf*2.0f - 1.0f, -1.0f), 1.0f);
    float gy = fminf(fmaxf(ay/pf*2.0f - 1.0f, -1.0f), 1.0f);
    grid32[((size_t)n*2 + 0)*1024 + it] = gx;
    grid32[((size_t)n*2 + 1)*1024 + it] = gy;
  }
}

// ---------------------------------------------------------------------------
// K5: bilinear upsample grid 32->224 (align_corners) fused with grid_sample
// on the full-res 256x256 frames; writes [B,C,T,224,224]. Gathers stay in
// the 192 MB L2 (whole input fits).
// ---------------------------------------------------------------------------
__global__ __launch_bounds__(256) void k_sample(const float* __restrict__ x,
    const float* __restrict__ grid32, float* __restrict__ out){
  const int img = blockIdx.x / 196;
  const int pix = (blockIdx.x % 196)*256 + threadIdx.x;   // 224*224 = 196*256
  const int oy = pix / 224, ox = pix % 224;
  const int b = img >> 5, t = img & 31;
  const float* g = grid32 + (size_t)img*2048;
  const float sc = 31.0f/223.0f;
  float fy = oy*sc, fx = ox*sc;
  int y0 = (int)fy, x0 = (int)fx;
  int y1 = min(y0+1, 31), x1 = min(x0+1, 31);
  float wy = fy - y0, wx = fx - x0;
  float w00 = (1.0f-wx)*(1.0f-wy), w01 = wx*(1.0f-wy);
  float w10 = (1.0f-wx)*wy,        w11 = wx*wy;
  int i00 = y0*32+x0, i01 = y0*32+x1, i10 = y1*32+x0, i11 = y1*32+x1;
  float gxv = g[i00]*w00 + g[i01]*w01 + g[i10]*w10 + g[i11]*w11;
  float gyv = g[1024+i00]*w00 + g[1024+i01]*w01 + g[1024+i10]*w10 + g[1024+i11]*w11;
  float ix = (gxv + 1.0f)*0.5f*255.0f;
  float iy = (gyv + 1.0f)*0.5f*255.0f;
  int ix0 = iclamp((int)floorf(ix), 0, 255);
  int iy0 = iclamp((int)floorf(iy), 0, 255);
  int ix1 = min(ix0+1, 255), iy1 = min(iy0+1, 255);
  float sx = ix - ix0, sy = iy - iy0;
  float s00 = (1.0f-sx)*(1.0f-sy), s01 = sx*(1.0f-sy);
  float s10 = (1.0f-sx)*sy,        s11 = sx*sy;
  #pragma unroll
  for (int c = 0; c < 3; ++c){
    const float* im = x + (((size_t)(b*3 + c))*32 + t)*65536;
    float v = im[iy0*256+ix0]*s00 + im[iy0*256+ix1]*s01
            + im[iy1*256+ix0]*s10 + im[iy1*256+ix1]*s11;
    out[(((size_t)(b*3 + c))*32 + t)*50176 + pix] = v;
  }
}

extern "C" void kernel_launch(void* const* d_in, const int* in_sizes, int n_in,
                              void* d_out, int out_size, void* d_ws, size_t ws_size,
                              hipStream_t stream){
  (void)in_sizes; (void)n_in; (void)out_size; (void)ws_size;
  const float* x     = (const float*)d_in[0];
  // d_in[1] = p (int scalar): p==1 deterministically skips the random branch.
  const float* sal_w = (const float*)d_in[2];
  const float* sal_b = (const float*)d_in[3];
  const float* cl_w  = (const float*)d_in[4];
  const float* cl_b  = (const float*)d_in[5];
  const float* gauss = (const float*)d_in[6];
  float* out = (float*)d_out;

  char* ws = (char*)d_ws;
  _Float16* xlow   = (_Float16*)(ws);                            // 12,582,912 B
  float*    xs     = (float*)(ws + 12582912);                    //  8,388,608 B
  float*    hm     = (float*)(ws + 12582912 + 8388608);          //    524,288 B
  float*    grid32 = (float*)(ws + 12582912 + 8388608 + 524288); //  1,048,576 B
  _Float16* bfrag  = (_Float16*)(ws + 12582912 + 8388608 + 524288 + 1048576); // 4,096 B

  k_prep_b        <<<8,       256, 0, stream>>>(sal_w, bfrag);
  k_pool          <<<24576,   256, 0, stream>>>(x, xlow);
  k_conv_wmma     <<<8192,    256, 0, stream>>>(xlow, bfrag, sal_b, cl_w, cl_b, xs);
  k_resize_softmax<<<128,     256, 0, stream>>>(xs, hm);
  k_gauss_grid    <<<128,     256, 0, stream>>>(hm, gauss, grid32);
  k_sample        <<<128*196, 256, 0, stream>>>(x, grid32, out);
}
